// LSTM_6356551598445
// MI455X (gfx1250) — matrix operand order repaired and checked by reading, not compiled, not run on hardware
//
#include <hip/hip_runtime.h>

// LSTM fused kernel for MI455X (gfx1250, wave32, WMMA bf16 16x16x32).
//
// Grid: 16 blocks (16 batch rows each), 512 threads (16 waves).
// Each wave owns 32 output-gate columns; W_ih/W_hh live in registers as
// pre-converted bf16 WMMA B-fragments. h state is bf16 in LDS (A-fragments),
// c state is in registers. x tiles are double-buffered through LDS.

typedef __attribute__((ext_vector_type(16))) __bf16 v16bf;
typedef __attribute__((ext_vector_type(8)))  float  v8f;

#define T_STEPS 1024
#define NIN     64
#define NH      128
#define NG      512        // 4*NH, PyTorch gate order i,f,g,o
#define ROWS    16         // batch rows per block
#define THREADS 512        // 16 waves
#define NBLK    16         // 256 / ROWS

union FragBF {
    v16bf v;
    unsigned short s[16];
    uint4 q2[2];
};

__device__ __forceinline__ unsigned short f2bf(float f) {
    union { float f; unsigned u; } x; x.f = f;
    unsigned r = x.u + 0x7FFFu + ((x.u >> 16) & 1u);   // round-to-nearest-even
    return (unsigned short)(r >> 16);
}

__device__ __forceinline__ float sigm(float x) {
    return 1.0f / (1.0f + __expf(-x));
}
__device__ __forceinline__ float tanh_fast(float x) {
    float e = __expf(2.0f * x);
    return 1.0f - 2.0f / (e + 1.0f);     // -> +/-1 correctly at saturation
}

__global__ __launch_bounds__(THREADS) void lstm_fused_kernel(
    const float* __restrict__ x,      // [256,1024,64]
    const float* __restrict__ W_ih,   // [512,64]
    const float* __restrict__ W_hh,   // [512,128]
    const float* __restrict__ b_ih,   // [512]
    const float* __restrict__ b_hh,   // [512]
    float* __restrict__ out)          // [256,1024,128]
{
    __shared__ __align__(16) unsigned short sX[2][ROWS][NIN];  // 4 KB  (bf16 x tiles)
    __shared__ __align__(16) unsigned short sH[ROWS][NH];      // 4 KB  (bf16 h state)
    __shared__ __align__(16) float          sG[ROWS][NG];      // 32 KB (f32 gate pre-acts)

    const int tid  = threadIdx.x;
    const int lane = tid & 31;
    const int wave = tid >> 5;
    const int b0   = blockIdx.x * ROWS;

    // ---------------- B-fragments (weights) -> registers, once ----------------
    // Wave covers gate columns [wave*32, wave*32+32) as two 16-col WMMA tiles.
    // B (32x16 KxN) as packed here (contiguous k): lane = column (mod 16); lanes 0-15 hold K=0..15,
    // lanes 16-31 hold K=16..31, 2 bf16 per VGPR.  Column n of B == row n of W
    // (since gates = h @ W^T), which is contiguous in memory.
    const int colA = wave * 32 + (lane & 15);
    const int colB = colA + 16;
    const int kb2  = (lane >> 4) << 4;            // 0 or 16

    FragBF Bih[2][2];                              // [k-chunk][tile]
#pragma unroll
    for (int kc = 0; kc < 2; ++kc)
#pragma unroll
        for (int nt = 0; nt < 2; ++nt) {
            const float* src = W_ih + (nt ? colB : colA) * NIN + kc * 32 + kb2;
#pragma unroll
            for (int j = 0; j < 16; ++j) Bih[kc][nt].s[j] = f2bf(src[j]);
        }

    FragBF Bhh[4][2];
#pragma unroll
    for (int kc = 0; kc < 4; ++kc)
#pragma unroll
        for (int nt = 0; nt < 2; ++nt) {
            const float* src = W_hh + (nt ? colB : colA) * NH + kc * 32 + kb2;
#pragma unroll
            for (int j = 0; j < 16; ++j) Bhh[kc][nt].s[j] = f2bf(src[j]);
        }

    // ---------------- elementwise mapping: 4 cells per thread ----------------
    const int eb = tid >> 5;            // batch row 0..15
    const int ej = (lane) * 4;          // hidden cols ej..ej+3
    float bi[4], bf[4], bg[4], bo[4];
#pragma unroll
    for (int u = 0; u < 4; ++u) {
        bi[u] = b_ih[       ej + u] + b_hh[       ej + u];
        bf[u] = b_ih[  NH + ej + u] + b_hh[  NH + ej + u];
        bg[u] = b_ih[2*NH + ej + u] + b_hh[2*NH + ej + u];
        bo[u] = b_ih[3*NH + ej + u] + b_hh[3*NH + ej + u];
    }
    float c_st[4] = {0.f, 0.f, 0.f, 0.f};

    // zero initial h (all threads, no divergence: 2048/512 = 4 each)
#pragma unroll
    for (int i = tid; i < ROWS * NH; i += THREADS)
        (&sH[0][0])[i] = (unsigned short)0;

    // ---------------- x tile prefetch for t = 0 ----------------
    // tile = 16*64 = 1024 f32; 512 threads -> float2 each
    const int pidx = tid * 2;
    const int pb   = pidx >> 6;
    const int pk   = pidx & 63;
    const float* xrow = x + (long)(b0 + pb) * T_STEPS * NIN + pk;
    {
        float2 v = *(const float2*)xrow;
        sX[0][pb][pk]     = f2bf(v.x);
        sX[0][pb][pk + 1] = f2bf(v.y);
    }
    __syncthreads();

    // A (16x32 MxK) as loaded here (contiguous k, same as B): lane%16 = row M; lanes 0-15 hold K 0-15,
    // lanes 16-31 hold K 16-31 (two 16-byte ds_load_b128 per chunk).
    const int arow = lane & 15;
    const int akh  = (lane >> 4) * 16;  // 0 or 16
    const int m0   = (lane >> 4) * 8;   // D-tile M offset for this lane half

    float* outp = out + ((long)(b0 + eb) * T_STEPS) * NH + ej;

    for (int t = 0; t < T_STEPS; ++t) {
        const int cur = t & 1, nxt = cur ^ 1;

        // issue next x-tile global load early (clamped, branch-free)
        const int tn = (t + 1 < T_STEPS) ? (t + 1) : t;
        float2 xv = *(const float2*)(xrow + (long)tn * NIN);

        // ---- A fragments from LDS ----
        FragBF Ax[2], Ah[4];
#pragma unroll
        for (int kc = 0; kc < 2; ++kc) {
            const unsigned short* p = &sX[cur][arow][kc * 32 + akh];
            Ax[kc].q2[0] = *(const uint4*)p;
            Ax[kc].q2[1] = *(const uint4*)(p + 8);
        }
#pragma unroll
        for (int kc = 0; kc < 4; ++kc) {
            const unsigned short* p = &sH[arow][kc * 32 + akh];
            Ah[kc].q2[0] = *(const uint4*)p;
            Ah[kc].q2[1] = *(const uint4*)(p + 8);
        }

        // ---- 12 WMMAs: gates tile [16 x 32] for this wave ----
        v8f acc0 = {}, acc1 = {};
#pragma unroll
        for (int kc = 0; kc < 2; ++kc) {
            acc0 = __builtin_amdgcn_wmma_f32_16x16x32_bf16(false, Ax[kc].v, false, Bih[kc][0].v, (short)0, acc0, false, false);
            acc1 = __builtin_amdgcn_wmma_f32_16x16x32_bf16(false, Ax[kc].v, false, Bih[kc][1].v, (short)0, acc1, false, false);
        }
#pragma unroll
        for (int kc = 0; kc < 4; ++kc) {
            acc0 = __builtin_amdgcn_wmma_f32_16x16x32_bf16(false, Ah[kc].v, false, Bhh[kc][0].v, (short)0, acc0, false, false);
            acc1 = __builtin_amdgcn_wmma_f32_16x16x32_bf16(false, Ah[kc].v, false, Bhh[kc][1].v, (short)0, acc1, false, false);
        }

        // ---- stage gate pre-activations (D layout: vgpr r -> M = m0+r) ----
#pragma unroll
        for (int r = 0; r < 8; ++r) {
            sG[m0 + r][colA] = acc0[r];
            sG[m0 + r][colB] = acc1[r];
        }

        // land prefetched x tile into the other buffer
        sX[nxt][pb][pk]     = f2bf(xv.x);
        sX[nxt][pb][pk + 1] = f2bf(xv.y);

        __syncthreads();

        // ---- elementwise LSTM cell update (4 cells / thread) ----
#pragma unroll
        for (int u = 0; u < 4; ++u) {
            float gi = sigm     (sG[eb][       ej + u] + bi[u]);
            float gf = sigm     (sG[eb][  NH + ej + u] + bf[u]);
            float gg = tanh_fast(sG[eb][2*NH + ej + u] + bg[u]);
            float go = sigm     (sG[eb][3*NH + ej + u] + bo[u]);
            float c  = gf * c_st[u] + gi * gg;
            c_st[u]  = c;
            float h  = go * tanh_fast(c);
            sH[eb][ej + u] = f2bf(h);
            outp[u] = h;
        }
        outp += NH;

        __syncthreads();
    }
}

extern "C" void kernel_launch(void* const* d_in, const int* in_sizes, int n_in,
                              void* d_out, int out_size, void* d_ws, size_t ws_size,
                              hipStream_t stream) {
    const float* x    = (const float*)d_in[0];
    const float* W_ih = (const float*)d_in[1];
    const float* W_hh = (const float*)d_in[2];
    const float* b_ih = (const float*)d_in[3];
    const float* b_hh = (const float*)d_in[4];
    float* out = (float*)d_out;

    lstm_fused_kernel<<<dim3(NBLK), dim3(THREADS), 0, stream>>>(
        x, W_ih, W_hh, b_ih, b_hh, out);
}
